// NonAutoRegressive_89730456748095
// MI455X (gfx1250) — matrix operand order repaired and checked by reading, not compiled, not run on hardware
//
#include <hip/hip_runtime.h>
#include <hip/hip_bf16.h>

// GatedGCN (N=50000, E=400000, D=64, 4 layers) for gfx1250 (MI455X).
// GEMMs on v_wmma_f32_16x16x32_bf16; LDS tiles stored in WMMA fragment order
// so fragment loads are 2x ds_load_b128; contiguous tile staging via the
// Tensor Data Mover (tensor_load_to_lds + s_wait_tensorcnt) where available.

#define DIM      64
#define TPOS     113          // L - K + 1 = 128 - 16 + 1
#define EPS_BN   1e-5f
#define EPS_AGG  1e-6f
#define NLAYERS  4

typedef __attribute__((ext_vector_type(16))) __bf16 bf16x16;
typedef __attribute__((ext_vector_type(8)))  float  floatx8;
typedef __attribute__((ext_vector_type(4)))  unsigned int u32x4;
typedef __attribute__((ext_vector_type(8)))  int          i32x8;
typedef __attribute__((ext_vector_type(4)))  int          i32x4;
typedef unsigned short u16;

// ---------- bf16 helpers -----------------------------------------------------
__device__ __forceinline__ u16 f2bfbits(float f) {
  union { float f; unsigned int u; } v; v.f = f;
  unsigned int u = v.u;
  unsigned int r = u + 0x7FFFu + ((u >> 16) & 1u);   // RNE
  return (u16)(r >> 16);
}

// ---------- WMMA wrapper -----------------------------------------------------
__device__ __forceinline__ floatx8 wmma_bf16(bf16x16 a, bf16x16 b, floatx8 c) {
  return __builtin_amdgcn_wmma_f32_16x16x32_bf16(false, a, false, b, (short)0, c,
                                                 false, false);
}
__device__ __forceinline__ bf16x16 frag_ld(const u16* p) {
  return *reinterpret_cast<const bf16x16*>(p);
}

// ---------- fragment-order addressing (CDNA5 ISA 7.12.2 layouts) -------------
// One fragment = 512 u16 = [lane 0..31][elem 0..15], 32B per lane.
// A 16x32 bf16, lane = mrow + 16*half:
//   kk 0..7  -> half 0, i = kk        kk 8..15  -> half 1, i = kk-8
//   kk 16..23-> half 0, i = kk-8      kk 24..31 -> half 1, i = kk-16
__device__ __forceinline__ int fragA_off(int mrow, int kk) {
  int half = (kk >> 3) & 1;
  int i = kk - (half ? 8 : 0) - ((kk >= 16) ? 8 : 0);
  return (mrow + 16 * half) * 16 + i;
}
// B 32x16 bf16, lane = n + 16*((kk>>3)&1), elem i = (kk & 7) + 8*(kk>>4)
__device__ __forceinline__ int fragB_off(int n, int kk) {
  return (n + 16 * ((kk >> 3) & 1)) * 16 + (kk & 7) + ((kk >> 4) << 3);
}
// global A-swizzled buffers: [rowtile][kstep(2)][512]
__device__ __forceinline__ size_t swizA_glob(size_t row, int col) {
  return ((row >> 4) * 2 + (size_t)(col >> 5)) * 512 +
         (size_t)fragA_off((int)(row & 15), col & 31);
}
// C/D 16x16 f32: VGPR r holds (m = r + 8*half, n = lane&15)

// ---------- TDM: contiguous global -> LDS copy (1-D tile descriptor) ---------
__device__ __forceinline__ void tdm_load_to_lds(const void* gsrc, void* lds_dst,
                                                unsigned bytes) {
#if __has_builtin(__builtin_amdgcn_tensor_load_to_lds)
  if (threadIdx.x < 32) {                     // one wave issues the DMA
    unsigned lds_off = (unsigned)(size_t)lds_dst;
    unsigned long long ga = (unsigned long long)(size_t)gsrc;
    unsigned n8 = bytes >> 3;                 // 8-byte elements
    u32x4 g0; i32x8 g1; i32x4 g2 = {0, 0, 0, 0}, g3 = {0, 0, 0, 0};
    g0[0] = 1u;                                             // count=1, user D#
    g0[1] = lds_off;                                        // lds_addr
    g0[2] = (unsigned)ga;                                   // global_addr lo
    g0[3] = (unsigned)((ga >> 32) & 0x1FFFFFFu) | (2u << 30); // addr hi | type=2
    g1[0] = (int)(3u << 16);                                // data_size=8B
    g1[1] = (int)((n8 & 0xFFFFu) << 16);                    // tensor_dim0 lo16
    g1[2] = (int)((n8 >> 16) | (1u << 16));                 // dim0 hi | dim1=1
    g1[3] = (int)((n8 & 0xFFFFu) << 16);                    // tile_dim0
    g1[4] = 1;                                              // tile_dim1=1
    g1[5] = (int)n8;                                        // dim0_stride lo
    g1[6] = 0; g1[7] = 0;
#if __has_include(<hip/amd_detail/amd_gfx1250_TDM.h>)
    i32x8 g4 = {0, 0, 0, 0, 0, 0, 0, 0};
    __builtin_amdgcn_tensor_load_to_lds(g0, g1, g2, g3, g4, 0);
#else
    __builtin_amdgcn_tensor_load_to_lds(g0, g1, g2, g3, 0);
#endif
    __builtin_amdgcn_s_wait_tensorcnt((short)0);
  }
  __syncthreads();
#else
  const uint4* s = (const uint4*)gsrc;
  uint4* d = (uint4*)lds_dst;
  for (unsigned i = threadIdx.x; i < (bytes >> 4); i += blockDim.x) d[i] = s[i];
  __syncthreads();
#endif
}

// ---------- weight conversion: f32 [K][64] (or transposed) -> fragment order -
// out frag id = nt*(K/32) + s, frag = 512 u16.
__global__ void swizzleB_cvt_kernel(const float* __restrict__ in,
                                    u16* __restrict__ out, int K, int trans) {
  int idx = blockIdx.x * 256 + threadIdx.x;
  if (idx >= K * 64) return;
  int k = idx >> 6, d = idx & 63;
  float v = trans ? in[d * K + k] : in[k * 64 + d];
  int nt = d >> 4, n = d & 15, s = k >> 5, kk = k & 31;
  out[(size_t)(nt * (K / 32) + s) * 512 + fragB_off(n, kk)] = f2bfbits(v);
}

// ---------- SequenceEncoder: conv1d(4->64,K=16) + maxpool --------------------
__global__ void __launch_bounds__(128)
seq_conv_kernel(const float* __restrict__ reads, const u16* __restrict__ conv_wbuf,
                const float* __restrict__ conv_b, float* __restrict__ h,
                u16* __restrict__ h_bf, u16* __restrict__ h_bfA, int N) {
  __shared__ __align__(64) u16 xs[16 * 512];   // 16 A-frags (8 mtiles x 2 ksteps)
  __shared__ __align__(64) u16 wl[8 * 512];    // 8 B-frags (4 ntiles x 2 ksteps)
  const int n = blockIdx.x;
  const int tid = threadIdx.x, lane = tid & 31, wave = tid >> 5, hf = (tid & 31) >> 4;
  const float* rb = reads + (size_t)n * 512;   // [4][128]
  tdm_load_to_lds(conv_wbuf, wl, 8 * 1024);
  for (int idx = tid; idx < 128 * 64; idx += 128) {
    int t = idx >> 6, k = idx & 63;
    int c = k >> 4, tap = k & 15;
    float v = (t < TPOS) ? rb[c * 128 + t + tap] : 0.f;
    int mt = t >> 4, mrow = t & 15, s = k >> 5, kk = k & 31;
    xs[(mt * 2 + s) * 512 + fragA_off(mrow, kk)] = f2bfbits(v);
  }
  __syncthreads();
  bf16x16 b0 = frag_ld(&wl[(wave * 2 + 0) * 512 + lane * 16]);
  bf16x16 b1 = frag_ld(&wl[(wave * 2 + 1) * 512 + lane * 16]);
  float lanemax = -1e30f;
  for (int mt = 0; mt < 8; ++mt) {
    bf16x16 a0 = frag_ld(&xs[(mt * 2 + 0) * 512 + lane * 16]);
    bf16x16 a1 = frag_ld(&xs[(mt * 2 + 1) * 512 + lane * 16]);
    floatx8 acc = {0.f, 0.f, 0.f, 0.f, 0.f, 0.f, 0.f, 0.f};
    acc = wmma_bf16(a0, b0, acc);
    acc = wmma_bf16(a1, b1, acc);
#pragma unroll
    for (int r = 0; r < 8; ++r) {
      int t = mt * 16 + r + 8 * hf;
      lanemax = fmaxf(lanemax, (t < TPOS) ? acc[r] : -1e30f);
    }
  }
  lanemax = fmaxf(lanemax, __shfl_xor(lanemax, 16, 32));
  if (lane < 16) {
    int ch = wave * 16 + lane;
    float o = lanemax + conv_b[ch];
    h[(size_t)n * DIM + ch] = o;
    u16 ob = f2bfbits(o);
    h_bf[(size_t)n * DIM + ch] = ob;
    h_bfA[swizA_glob((size_t)n, ch)] = ob;
  }
}

// ---------- EdgeEncoder ------------------------------------------------------
__global__ void edge_enc_kernel(const float* __restrict__ sim,
                                const float* __restrict__ len,
                                const float* __restrict__ We,
                                const float* __restrict__ be,
                                float* __restrict__ e, u16* __restrict__ e_bfA,
                                size_t total) {
  size_t i = (size_t)blockIdx.x * blockDim.x + threadIdx.x;
  size_t stride = (size_t)gridDim.x * blockDim.x;
  for (; i < total; i += stride) {
    int c = (int)(i & 63);
    size_t ed = i >> 6;
    float v = sim[ed] * We[c] + len[ed] * We[64 + c] + be[c];
    e[i] = v;
    e_bfA[swizA_glob(ed, c)] = f2bfbits(v);
  }
}

// ---------- Node projections: HA1, HA2, HV, HU (one 16-row tile / block) -----
__global__ void __launch_bounds__(128)
node_gemm_kernel(const u16* __restrict__ h_bfA, const u16* __restrict__ wnode,
                 const float* __restrict__ bi0, const float* __restrict__ bi1,
                 const float* __restrict__ bi2, const float* __restrict__ bi3,
                 float* __restrict__ o0, float* __restrict__ o1,
                 float* __restrict__ o2, float* __restrict__ o3, int N) {
  __shared__ __align__(64) u16 hl[2 * 512];    // 2 A-frags (16 rows, K=64)
  __shared__ __align__(64) u16 wl[4 * 8 * 512];// 4 matrices x 8 B-frags
  int tid = threadIdx.x, lane = tid & 31, wave = tid >> 5, hf = (tid & 31) >> 4;
  size_t rowbase = (size_t)blockIdx.x * 16;
  tdm_load_to_lds(wnode, wl, 4 * 8 * 1024);
  tdm_load_to_lds(h_bfA + (size_t)blockIdx.x * 1024, hl, 2 * 1024);
  bf16x16 a0 = frag_ld(&hl[0 * 512 + lane * 16]);
  bf16x16 a1 = frag_ld(&hl[1 * 512 + lane * 16]);
  const float* bias = (wave == 0) ? bi0 : (wave == 1) ? bi1 : (wave == 2) ? bi2 : bi3;
  float* out = (wave == 0) ? o0 : (wave == 1) ? o1 : (wave == 2) ? o2 : o3;
  const u16* wbase = wl + wave * 8 * 512;
  for (int nt = 0; nt < 4; ++nt) {
    bf16x16 bb0 = frag_ld(&wbase[(nt * 2 + 0) * 512 + lane * 16]);
    bf16x16 bb1 = frag_ld(&wbase[(nt * 2 + 1) * 512 + lane * 16]);
    floatx8 acc = {0.f, 0.f, 0.f, 0.f, 0.f, 0.f, 0.f, 0.f};
    acc = wmma_bf16(a0, bb0, acc);
    acc = wmma_bf16(a1, bb1, acc);
    int ch = nt * 16 + (lane & 15);
    float bv = bias[ch];
#pragma unroll
    for (int r = 0; r < 8; ++r) {
      size_t node = rowbase + r + 8 * hf;
      if (node < (size_t)N) out[node * DIM + ch] = acc[r] + bv;
    }
  }
}

// ---------- Fused edge pass: e_hat = e@A3+bA3 + HA1[src] + HA2[dst];
//            sigma = sigmoid(e_hat); atomic scatter num/den to dst. ----------
__global__ void __launch_bounds__(128)
edge_gemm_kernel(const u16* __restrict__ e_bfA, const u16* __restrict__ wA3buf,
                 const float* __restrict__ bA3, const int* __restrict__ src,
                 const int* __restrict__ dst, const float* __restrict__ HA1,
                 const float* __restrict__ HA2, const float* __restrict__ HV,
                 float* __restrict__ e_hat, float* __restrict__ numb,
                 float* __restrict__ den, int E) {
  __shared__ __align__(64) u16 el[8 * 512];    // 4 edge-tiles x 2 ksteps
  __shared__ __align__(64) u16 wl[8 * 512];    // A3: 4 ntiles x 2 ksteps
  int tid = threadIdx.x, lane = tid & 31, wave = tid >> 5, hf = (tid & 31) >> 4;
  size_t ebase = (size_t)blockIdx.x * 64;
  tdm_load_to_lds(wA3buf, wl, 8 * 1024);
  tdm_load_to_lds(e_bfA + (size_t)blockIdx.x * 4096, el, 8 * 1024);
  {  // prefetch upcoming gather lines (global_prefetch_b8)
    size_t ed = ebase + wave * 16 + (lane & 15);
    if (ed < (size_t)E) {
      __builtin_prefetch(&HV[(size_t)src[ed] * DIM], 0, 1);
      __builtin_prefetch(&HA2[(size_t)dst[ed] * DIM], 0, 1);
    }
  }
  bf16x16 a0 = frag_ld(&el[(wave * 2 + 0) * 512 + lane * 16]);
  bf16x16 a1 = frag_ld(&el[(wave * 2 + 1) * 512 + lane * 16]);
  for (int nt = 0; nt < 4; ++nt) {
    bf16x16 bb0 = frag_ld(&wl[(nt * 2 + 0) * 512 + lane * 16]);
    bf16x16 bb1 = frag_ld(&wl[(nt * 2 + 1) * 512 + lane * 16]);
    floatx8 acc = {0.f, 0.f, 0.f, 0.f, 0.f, 0.f, 0.f, 0.f};
    acc = wmma_bf16(a0, bb0, acc);
    acc = wmma_bf16(a1, bb1, acc);
    int n = nt * 16 + (lane & 15);
    float bv = bA3[n];
#pragma unroll
    for (int r = 0; r < 8; ++r) {
      size_t ed = ebase + wave * 16 + r + 8 * hf;
      if (ed < (size_t)E) {
        int s = src[ed], d2 = dst[ed];
        float eh = acc[r] + bv + HA1[(size_t)s * DIM + n] + HA2[(size_t)d2 * DIM + n];
        e_hat[ed * DIM + n] = eh;
        float sg = 1.f / (1.f + __expf(-eh));
        atomicAdd(&den[(size_t)d2 * DIM + n], sg);
        atomicAdd(&numb[(size_t)d2 * DIM + n], sg * HV[(size_t)s * DIM + n]);
      }
    }
  }
}

// ---------- h_hat = HU + num/(den+eps) ---------------------------------------
__global__ void hhat_kernel(const float* __restrict__ HU, const float* __restrict__ numb,
                            const float* __restrict__ den, float* __restrict__ h_hat,
                            size_t total) {
  size_t i = (size_t)blockIdx.x * blockDim.x + threadIdx.x;
  size_t stride = (size_t)gridDim.x * blockDim.x;
  for (; i < total; i += stride) h_hat[i] = HU[i] + numb[i] / (den[i] + EPS_AGG);
}

// ---------- per-channel BN stats (sum, sumsq) --------------------------------
__global__ void __launch_bounds__(256)
stats_kernel(const float* __restrict__ x, size_t rows, float* __restrict__ stats) {
  __shared__ float ss[256], sq[256];
  int tid = threadIdx.x;
  int c = tid & 63;
  size_t total = rows * DIM;
  size_t i = (size_t)blockIdx.x * 256 + tid;
  size_t stride = (size_t)gridDim.x * 256;
  float s = 0.f, q = 0.f;
  for (; i < total; i += stride) { float v = x[i]; s += v; q += v * v; }
  ss[tid] = s; sq[tid] = q;
  __syncthreads();
  if (tid < 64) {
    atomicAdd(&stats[c], ss[tid] + ss[tid + 64] + ss[tid + 128] + ss[tid + 192]);
    atomicAdd(&stats[64 + c], sq[tid] + sq[tid + 64] + sq[tid + 128] + sq[tid + 192]);
  }
}

// ---------- x = x + relu(BN(xhat)); refresh bf16 copies ----------------------
__global__ void bn_apply_kernel(const float* __restrict__ xhat,
                                const float* __restrict__ stats,
                                const float* __restrict__ gamma,
                                const float* __restrict__ beta, float invR,
                                float* __restrict__ x, u16* __restrict__ x_bf_row,
                                u16* __restrict__ x_bfA, size_t total) {
  size_t i = (size_t)blockIdx.x * blockDim.x + threadIdx.x;
  size_t stride = (size_t)gridDim.x * blockDim.x;
  for (; i < total; i += stride) {
    int c = (int)(i & 63);
    float mean = stats[c] * invR;
    float var = stats[64 + c] * invR - mean * mean;
    float v = gamma[c] * (xhat[i] - mean) * rsqrtf(var + EPS_BN) + beta[c];
    float nv = x[i] + fmaxf(v, 0.f);
    x[i] = nv;
    u16 b = f2bfbits(nv);
    if (x_bf_row) x_bf_row[i] = b;
    x_bfA[swizA_glob(i >> 6, c)] = b;
  }
}

// ---------- decoder: relu([h_s|h_d|e] @ W1 + b1) @ W2 + b2 -------------------
__global__ void __launch_bounds__(128)
decoder_kernel(const u16* __restrict__ h_bf, const u16* __restrict__ e_bfA,
               const int* __restrict__ src, const int* __restrict__ dst,
               const u16* __restrict__ w1buf, const float* __restrict__ b1,
               const float* __restrict__ W2, const float* __restrict__ b2,
               float* __restrict__ out, int E) {
  __shared__ __align__(64) u16 zl[24 * 512];   // A-frags: [ks 0..5][wave 0..3]
  __shared__ __align__(64) u16 wl[24 * 512];   // W1 B-frags: [nt 0..3][ks 0..5]
  int tid = threadIdx.x, lane = tid & 31, wave = tid >> 5, hf = (tid & 31) >> 4;
  size_t ebase = (size_t)blockIdx.x * 64;
  tdm_load_to_lds(w1buf, wl, 24 * 1024);
  // h_src | h_dst part: per-edge gather, swizzled into fragment order
  for (int idx = tid; idx < 64 * 128; idx += 128) {
    int m = idx >> 7, c = idx & 127;
    int w = m >> 4, mrow = m & 15, ks = c >> 5, kk = c & 31;
    size_t ed = ebase + m; if (ed >= (size_t)E) ed = (size_t)E - 1;
    u16 v = (c < 64) ? h_bf[(size_t)src[ed] * DIM + c]
                     : h_bf[(size_t)dst[ed] * DIM + (c - 64)];
    zl[(ks * 4 + w) * 512 + fragA_off(mrow, kk)] = v;
  }
  // e part: already fragment-ordered in e_bfA, plain remapped copy
  for (int idx = tid; idx < 8 * 64; idx += 128) {
    int f = idx >> 6, q = idx & 63;            // f = w*2+s, 64 x uint4 per frag
    int w = f >> 1, s = f & 1;
    const uint4* sp = (const uint4*)(e_bfA + ((size_t)blockIdx.x * 4 + w) * 1024 +
                                     (size_t)s * 512);
    uint4* dp = (uint4*)(zl + (size_t)((4 + s) * 4 + w) * 512);
    dp[q] = sp[q];
  }
  __syncthreads();
  bf16x16 af[6];
#pragma unroll
  for (int ks = 0; ks < 6; ++ks)
    af[ks] = frag_ld(&zl[(ks * 4 + wave) * 512 + lane * 16]);
  float accm[8];
#pragma unroll
  for (int r = 0; r < 8; ++r) accm[r] = 0.f;
  for (int nt = 0; nt < 4; ++nt) {
    floatx8 acc = {0.f, 0.f, 0.f, 0.f, 0.f, 0.f, 0.f, 0.f};
#pragma unroll
    for (int ks = 0; ks < 6; ++ks) {
      bf16x16 bb = frag_ld(&wl[(nt * 6 + ks) * 512 + lane * 16]);
      acc = wmma_bf16(af[ks], bb, acc);
    }
    int n = nt * 16 + (lane & 15);
    float w2v = W2[n], b1v = b1[n];
#pragma unroll
    for (int r = 0; r < 8; ++r) {
      float v = fmaxf(acc[r] + b1v, 0.f) * w2v;
      v += __shfl_xor(v, 1, 32);
      v += __shfl_xor(v, 2, 32);
      v += __shfl_xor(v, 4, 32);
      v += __shfl_xor(v, 8, 32);
      accm[r] += v;
    }
  }
  if ((lane & 15) == 0) {
#pragma unroll
    for (int r = 0; r < 8; ++r) {
      size_t ed = ebase + wave * 16 + r + 8 * hf;
      if (ed < (size_t)E) out[ed] = accm[r] + b2[0];
    }
  }
}

// ============================ host side ======================================
extern "C" void kernel_launch(void* const* d_in, const int* in_sizes, int n_in,
                              void* d_out, int out_size, void* d_ws, size_t ws_size,
                              hipStream_t stream) {
  (void)n_in; (void)out_size; (void)ws_size;
  const float* reads  = (const float*)d_in[0];
  const int*   src    = (const int*)  d_in[1];
  const int*   dst    = (const int*)  d_in[2];
  const float* sim    = (const float*)d_in[3];
  const float* len    = (const float*)d_in[4];
  const float* conv_w = (const float*)d_in[5];
  const float* conv_b = (const float*)d_in[6];
  const float* We     = (const float*)d_in[7];
  const float* be     = (const float*)d_in[8];
  const float* A1     = (const float*)d_in[9];
  const float* A2     = (const float*)d_in[10];
  const float* A3     = (const float*)d_in[11];
  const float* bA1    = (const float*)d_in[12];
  const float* bA2    = (const float*)d_in[13];
  const float* bA3    = (const float*)d_in[14];
  const float* U      = (const float*)d_in[15];
  const float* V      = (const float*)d_in[16];
  const float* bU     = (const float*)d_in[17];
  const float* bV     = (const float*)d_in[18];
  const float* g_h    = (const float*)d_in[19];
  const float* b_h    = (const float*)d_in[20];
  const float* g_e    = (const float*)d_in[21];
  const float* b_e    = (const float*)d_in[22];
  const float* W1     = (const float*)d_in[23];
  const float* b1     = (const float*)d_in[24];
  const float* W2     = (const float*)d_in[25];
  const float* b2     = (const float*)d_in[26];
  float* out = (float*)d_out;

  const int N = in_sizes[0] / 512;
  const int E = in_sizes[1];
  const size_t N64 = (size_t)N * DIM, E64 = (size_t)E * DIM;
  const int ntiles = (N + 15) / 16, etiles64 = (E + 63) / 64;

  char* p = (char*)d_ws;
  auto alloc = [&](size_t bytes) -> void* {
    void* r = (void*)p;
    p += (bytes + 255) & ~(size_t)255;
    return r;
  };
  float* h     = (float*)alloc(N64 * 4);
  u16*   h_bf  = (u16*)  alloc(N64 * 2);                  // row-major (gathers)
  u16*   h_bfA = (u16*)  alloc((size_t)ntiles * 1024 * 2);// fragment order
  float* e     = (float*)alloc(E64 * 4);
  u16*   e_bfA = (u16*)  alloc((size_t)etiles64 * 4096 * 2);
  float* HA1   = (float*)alloc(N64 * 4);
  float* HA2   = (float*)alloc(N64 * 4);
  float* HV    = (float*)alloc(N64 * 4);
  float* HU    = (float*)alloc(N64 * 4);
  float* e_hat = (float*)alloc(E64 * 4);
  float* h_hat = (float*)alloc(N64 * 4);
  float* numb  = (float*)alloc(N64 * 4);
  float* den   = (float*)alloc(N64 * 4);
  float* st_h  = (float*)alloc(128 * 4);
  float* st_e  = (float*)alloc(128 * 4);
  u16* conv_wbuf = (u16*)alloc(4096 * 2);
  u16* wnode     = (u16*)alloc(4 * 4096 * 2);   // [A1|A2|V|U] fragment order
  u16* wA3buf    = (u16*)alloc(4096 * 2);
  u16* w1buf     = (u16*)alloc(192 * 64 * 2);

  // one-time weight swizzles
  swizzleB_cvt_kernel<<<16, 256, 0, stream>>>(conv_w, conv_wbuf, 64, 1);
  swizzleB_cvt_kernel<<<48, 256, 0, stream>>>(W1, w1buf, 192, 0);

  // encoders
  seq_conv_kernel<<<N, 128, 0, stream>>>(reads, conv_wbuf, conv_b, h, h_bf, h_bfA, N);
  edge_enc_kernel<<<2048, 256, 0, stream>>>(sim, len, We, be, e, e_bfA, E64);

  for (int l = 0; l < NLAYERS; ++l) {
    const size_t wo = (size_t)l * 4096, bo = (size_t)l * 64;
    swizzleB_cvt_kernel<<<16, 256, 0, stream>>>(A1 + wo, wnode + 0 * 4096, 64, 0);
    swizzleB_cvt_kernel<<<16, 256, 0, stream>>>(A2 + wo, wnode + 1 * 4096, 64, 0);
    swizzleB_cvt_kernel<<<16, 256, 0, stream>>>(V + wo, wnode + 2 * 4096, 64, 0);
    swizzleB_cvt_kernel<<<16, 256, 0, stream>>>(U + wo, wnode + 3 * 4096, 64, 0);
    swizzleB_cvt_kernel<<<16, 256, 0, stream>>>(A3 + wo, wA3buf, 64, 0);

    node_gemm_kernel<<<ntiles, 128, 0, stream>>>(
        h_bfA, wnode, bA1 + bo, bA2 + bo, bV + bo, bU + bo, HA1, HA2, HV, HU, N);

    hipMemsetAsync(numb, 0, N64 * 4, stream);
    hipMemsetAsync(den, 0, N64 * 4, stream);
    hipMemsetAsync(st_h, 0, 128 * 4, stream);
    hipMemsetAsync(st_e, 0, 128 * 4, stream);

    edge_gemm_kernel<<<etiles64, 128, 0, stream>>>(
        e_bfA, wA3buf, bA3 + bo, src, dst, HA1, HA2, HV, e_hat, numb, den, E);

    hhat_kernel<<<1024, 256, 0, stream>>>(HU, numb, den, h_hat, N64);
    stats_kernel<<<256, 256, 0, stream>>>(h_hat, (size_t)N, st_h);
    stats_kernel<<<512, 256, 0, stream>>>(e_hat, (size_t)E, st_e);
    bn_apply_kernel<<<1024, 256, 0, stream>>>(h_hat, st_h, g_h + bo, b_h + bo,
                                              1.f / (float)N, h, h_bf, h_bfA, N64);
    bn_apply_kernel<<<2048, 256, 0, stream>>>(e_hat, st_e, g_e + bo, b_e + bo,
                                              1.f / (float)E, e, nullptr, e_bfA, E64);
  }

  decoder_kernel<<<etiles64, 128, 0, stream>>>(h_bf, e_bfA, src, dst, w1buf, b1, W2,
                                               b2, out, E);
}
